// ImplicitHead2D_Seq_MS_Deform_49435073577581
// MI455X (gfx1250) — matrix operand order repaired and checked by reading, not compiled, not run on hardware
//
#include <hip/hip_runtime.h>
#include <hip/hip_bf16.h>

// ---------------------------------------------------------------------------
// MS-Deformable-Attention decoder layer for MI455X (gfx1250, wave32, WMMA).
// HBM-bound (~380MB traffic vs ~31 GFLOP): GEMMs use bf16 WMMA (fp32 accum),
// weights pre-packed once into a fragment-native bf16 layout so the hot loop
// is pure b128 loads + cvt_pk + v_wmma, software-pipelined over two k-chunks.
// ---------------------------------------------------------------------------

#define B_   8
#define NQ_  900
#define C_   256
#define M_   8
#define D_   32
#define P_   4
#define L_   4
#define S_   19560
#define FFN_ 1024
#define NC_  10
#define BS_  (B_ * S_)       // 156480 rows of the value projection
#define BQ_  (B_ * NQ_)      // 7200 query rows

typedef __attribute__((ext_vector_type(16))) __bf16 v16bf;
typedef __attribute__((ext_vector_type(8)))  __bf16 v8bf;
typedef __attribute__((ext_vector_type(8)))  float  v8f;

// ---------------------------------------------------------------------------
// Weight pre-pack: W[K x N] fp32 -> Wp[K/32][N][32] bf16.
// For a (32-wide k-chunk, column n) the 32 bf16 values sit in one 64B run, so
// a GEMM lane grabs its B fragment half with two b128 loads.
// ---------------------------------------------------------------------------
__global__ void pack_w_bf16(const float* __restrict__ W, __bf16* __restrict__ Wp,
                            int K, int N) {
  const int idx = blockIdx.x * 256 + threadIdx.x;   // index into Wp
  if (idx >= K * N) return;
  const int kk = idx & 31;
  const int t  = idx >> 5;      // kt*N + n
  const int n  = t % N;
  const int kt = t / N;
  Wp[idx] = (__bf16)W[(size_t)(kt * 32 + kk) * N + n];
}

// ---------------------------------------------------------------------------
// Fragment loaders.
// A 16x32 bf16 fragment (ISA 7.12.2): lane (row=l16, half) holds two
// contiguous 8-float runs: [k0+8*half, +8) and [k0+16+8*half, +8).
// B 32x16 fragment from packed layout: lane (col=l16, half) reads 16
// bf16 as two 8-runs at ((kt*N + n)*32 + 8*half) and +16 (same k order as A).
// ---------------------------------------------------------------------------
__device__ __forceinline__ v16bf load_a_frag(const float* __restrict__ xr,
                                             int k0, int half) {
  const float4 a0 = *(const float4*)(xr + k0 + 8 * half);
  const float4 a1 = *(const float4*)(xr + k0 + 8 * half + 4);
  const float4 a2 = *(const float4*)(xr + k0 + 16 + 8 * half);
  const float4 a3 = *(const float4*)(xr + k0 + 16 + 8 * half + 4);
  v16bf a;
  a[0]  = (__bf16)a0.x; a[1]  = (__bf16)a0.y; a[2]  = (__bf16)a0.z; a[3]  = (__bf16)a0.w;
  a[4]  = (__bf16)a1.x; a[5]  = (__bf16)a1.y; a[6]  = (__bf16)a1.z; a[7]  = (__bf16)a1.w;
  a[8]  = (__bf16)a2.x; a[9]  = (__bf16)a2.y; a[10] = (__bf16)a2.z; a[11] = (__bf16)a2.w;
  a[12] = (__bf16)a3.x; a[13] = (__bf16)a3.y; a[14] = (__bf16)a3.z; a[15] = (__bf16)a3.w;
  return a;
}

__device__ __forceinline__ v16bf load_b_frag(const __bf16* __restrict__ wp) {
  const v8bf lo = *(const v8bf*)wp;
  const v8bf hi = *(const v8bf*)(wp + 16);
  return __builtin_shufflevector(lo, hi, 0, 1, 2, 3, 4, 5, 6, 7,
                                 8, 9, 10, 11, 12, 13, 14, 15);
}

#define WMMA_BF16(A, Bf, Cacc)                                                 \
  __builtin_amdgcn_wmma_f32_16x16x32_bf16(false, (A), false, (Bf), (short)0,   \
                                          (Cacc), false, false)

// ---------------------------------------------------------------------------
// GEMM: Y[rows x N] = act(X[rows x K] @ W[K x N] + bias)
// One wave32 computes a 16x64 tile (4 accumulators, shared A fragments).
// Block = 256 threads = 8 waves; cgPerBlock column-groups x (8/cg) row-tiles.
// K % 64 == 0, N % 64 == 0 at every call site. k-loop unrolled over two
// 32-wide chunks: all 8 B fragments + 2 A fragments in flight per iteration
// so WMMAs overlap load latency instead of waiting per-fragment.
// ---------------------------------------------------------------------------
__global__ void gemm_bf16_wmma(const float* __restrict__ X,
                               const __bf16* __restrict__ Wp,  // packed
                               const float* __restrict__ bias,
                               float* __restrict__ Y,
                               int rows, int K, int N, int relu, int cgPerBlock) {
  const int lane = threadIdx.x & 31;
  const int wave = threadIdx.x >> 5;
  const int rt   = wave / cgPerBlock;                 // row tile within block
  const int cg   = wave % cgPerBlock;                 // column group within block
  const int r0   = (blockIdx.x * (8 / cgPerBlock) + rt) * 16;
  const int n0   = (blockIdx.y * cgPerBlock + cg) * 64;
  if (r0 >= rows || n0 >= N) return;

  const int half = lane >> 4;   // 0: lanes 0-15, 1: lanes 16-31
  const int l16  = lane & 15;

  const float* xr = X + (size_t)(r0 + l16) * K;       // A-fragment row
  // per-lane base into packed weights for this column group
  const __bf16* wbase = Wp + ((size_t)(n0 + l16)) * 32 + half * 8;

  v8f acc[4] = {{}, {}, {}, {}};

  for (int k0 = 0; k0 < K; k0 += 64) {
    if (k0 + 64 < K) __builtin_prefetch(xr + k0 + 64, 0, 3);  // global_prefetch_b8

    const v16bf a0 = load_a_frag(xr, k0, half);
    const v16bf a1 = load_a_frag(xr, k0 + 32, half);

    const __bf16* w0 = wbase + (size_t)(k0 >> 5) * N * 32;
    const __bf16* w1 = w0 + (size_t)N * 32;
    v16bf b0[4], b1[4];
#pragma unroll
    for (int j = 0; j < 4; ++j) b0[j] = load_b_frag(w0 + (size_t)j * 16 * 32);
#pragma unroll
    for (int j = 0; j < 4; ++j) b1[j] = load_b_frag(w1 + (size_t)j * 16 * 32);

#pragma unroll
    for (int j = 0; j < 4; ++j) acc[j] = WMMA_BF16(a0, b0[j], acc[j]);
#pragma unroll
    for (int j = 0; j < 4; ++j) acc[j] = WMMA_BF16(a1, b1[j], acc[j]);
  }

  // C/D layout: vgpr q -> row q (lanes 0-15) / row q+8 (lanes 16-31), col = lane&15
#pragma unroll
  for (int j = 0; j < 4; ++j) {
    const int c  = n0 + j * 16 + l16;
    const float bv = bias[c];
#pragma unroll
    for (int q = 0; q < 8; ++q) {
      const int r = r0 + q + half * 8;
      float v = acc[j][q] + bv;
      if (relu) v = fmaxf(v, 0.f);
      Y[(size_t)r * N + c] = v;
    }
  }
}

// ---------------------------------------------------------------------------
// Softmax over the 16 (L*P) logits per head. Block = 128 threads = one (b,q);
// thread t: m = t/16, i = t%16. Reductions via 16-wide shuffles inside wave32.
// ---------------------------------------------------------------------------
__global__ void softmax_lp(const float* __restrict__ logits,  // [BQ,128]
                           float* __restrict__ out) {         // [BQ,128]
  const int row = blockIdx.x;
  const int t   = threadIdx.x;          // 0..127
  float v = logits[(size_t)row * 128 + t];
  float mx = v;
#pragma unroll
  for (int d = 8; d >= 1; d >>= 1) mx = fmaxf(mx, __shfl_xor(mx, d, 16));
  float e = __expf(v - mx);
  float s = e;
#pragma unroll
  for (int d = 8; d >= 1; d >>= 1) s += __shfl_xor(s, d, 16);
  out[(size_t)row * 128 + t] = e / s;
}

// ---------------------------------------------------------------------------
// Deformable sampling. One wave per (b,q,m); lane = head channel d (exact
// wave32 fit). v layout [B,S,C] with c = m*32+d keeps a head's channels
// contiguous -> each corner gather is a coalesced 128B wave transaction,
// mostly hitting the 192MB L2 (v is 160MB).
// ---------------------------------------------------------------------------
__device__ __forceinline__ float corner(const float* __restrict__ vb, int st,
                                        int xi, int yi, int Hl, int Wl) {
  if (xi < 0 || xi >= Wl || yi < 0 || yi >= Hl) return 0.f;
  return vb[(size_t)(st + yi * Wl + xi) * C_];
}

__global__ void deform_sample(const float* __restrict__ v,     // [B,S,C]
                              const float* __restrict__ refp,  // [B,NQ,2]
                              const float* __restrict__ off,   // [BQ,256]
                              const float* __restrict__ attn,  // [BQ,128]
                              float* __restrict__ feat) {      // [BQ,C]
  const int Hs[L_] = {92, 46, 23, 12};
  const int Ws[L_] = {160, 80, 40, 20};
  const int St[L_] = {0, 14720, 18400, 19320};

  const int lane = threadIdx.x & 31;               // channel d
  const int wave = threadIdx.x >> 5;
  const int gid  = blockIdx.x * 8 + wave;          // (b,q,m) flat
  if (gid >= BQ_ * M_) return;
  const int m  = gid % M_;
  const int bq = gid / M_;
  const int b  = bq / NQ_;

  const float rx = refp[(size_t)bq * 2 + 0];
  const float ry = refp[(size_t)bq * 2 + 1];
  const float* vb = v + (size_t)b * S_ * C_ + m * D_ + lane;

  float acc = 0.f;
#pragma unroll
  for (int l = 0; l < L_; ++l) {
    const int Hl = Hs[l], Wl = Ws[l], st = St[l];
#pragma unroll
    for (int p = 0; p < P_; ++p) {
      const int oi = bq * 256 + (((m * L_ + l) * P_ + p) << 1);
      const float ox = off[oi + 0];
      const float oy = off[oi + 1];
      // loc = ref + off/[W,H]; grid coords: x*W-0.5, y*H-0.5
      const float fx = (rx + ox / (float)Wl) * (float)Wl - 0.5f;
      const float fy = (ry + oy / (float)Hl) * (float)Hl - 0.5f;
      const float x0f = floorf(fx), y0f = floorf(fy);
      const float wx = fx - x0f, wy = fy - y0f;
      const int x0 = (int)x0f, y0 = (int)y0f;
      const float a = attn[bq * 128 + m * 16 + l * P_ + p];

      const float s00 = corner(vb, st, x0,     y0,     Hl, Wl);
      const float s01 = corner(vb, st, x0 + 1, y0,     Hl, Wl);
      const float s10 = corner(vb, st, x0,     y0 + 1, Hl, Wl);
      const float s11 = corner(vb, st, x0 + 1, y0 + 1, Hl, Wl);
      const float s = s00 * (1.f - wx) * (1.f - wy) + s01 * wx * (1.f - wy)
                    + s10 * (1.f - wx) * wy         + s11 * wx * wy;
      acc = fmaf(a, s, acc);
    }
  }
  feat[(size_t)bq * C_ + m * D_ + lane] = acc;
}

// ---------------------------------------------------------------------------
// Fused residual-add + LayerNorm over 256-wide rows. One wave per row,
// 8 elements per lane, wave-wide shuffle reductions.
// ---------------------------------------------------------------------------
__global__ void add_layernorm(const float* __restrict__ xa,   // [rows,256]
                              const float* __restrict__ xb,   // [rows,256]
                              const float* __restrict__ w,    // [256]
                              const float* __restrict__ bi,   // [256]
                              float* __restrict__ y,          // [rows,256]
                              int rows) {
  const int lane = threadIdx.x & 31;
  const int wave = threadIdx.x >> 5;
  const int row  = blockIdx.x * 8 + wave;
  if (row >= rows) return;

  const float* pa = xa + (size_t)row * C_;
  const float* pb = xb + (size_t)row * C_;
  float vals[8];
  float s = 0.f, s2 = 0.f;
#pragma unroll
  for (int j = 0; j < 8; ++j) {
    const float t = pa[lane + 32 * j] + pb[lane + 32 * j];
    vals[j] = t;
    s  += t;
    s2 += t * t;
  }
#pragma unroll
  for (int d = 16; d >= 1; d >>= 1) {
    s  += __shfl_xor(s,  d, 32);
    s2 += __shfl_xor(s2, d, 32);
  }
  const float mean = s * (1.f / 256.f);
  const float var  = s2 * (1.f / 256.f) - mean * mean;
  const float inv  = rsqrtf(var + 1e-6f);
#pragma unroll
  for (int j = 0; j < 8; ++j) {
    const int c = lane + 32 * j;
    y[(size_t)row * C_ + c] = (vals[j] - mean) * inv * w[c] + bi[c];
  }
}

// ---------------------------------------------------------------------------
// Classification head: [BQ,256] @ [256,10] + b. Tiny (37 MFLOP) -> plain VALU.
// ---------------------------------------------------------------------------
__global__ void cls_head(const float* __restrict__ x, const float* __restrict__ w,
                         const float* __restrict__ b, float* __restrict__ out) {
  const int idx = blockIdx.x * blockDim.x + threadIdx.x;
  if (idx >= BQ_ * NC_) return;
  const int row = idx / NC_;
  const int c   = idx % NC_;
  float s = b[c];
  const float* xr = x + (size_t)row * C_;
#pragma unroll 8
  for (int k = 0; k < C_; ++k) s = fmaf(xr[k], w[k * NC_ + c], s);
  out[idx] = s;
}

// ---------------------------------------------------------------------------
extern "C" void kernel_launch(void* const* d_in, const int* in_sizes, int n_in,
                              void* d_out, int out_size, void* d_ws, size_t ws_size,
                              hipStream_t stream) {
  const float* query   = (const float*)d_in[0];
  const float* value   = (const float*)d_in[1];
  const float* refp    = (const float*)d_in[2];
  const float* w_value = (const float*)d_in[3];
  const float* b_value = (const float*)d_in[4];
  const float* w_off   = (const float*)d_in[5];
  const float* b_off   = (const float*)d_in[6];
  const float* w_attn  = (const float*)d_in[7];
  const float* b_attn  = (const float*)d_in[8];
  const float* w_out   = (const float*)d_in[9];
  const float* b_out   = (const float*)d_in[10];
  const float* ln1_w   = (const float*)d_in[11];
  const float* ln1_b   = (const float*)d_in[12];
  const float* w_ffn1  = (const float*)d_in[13];
  const float* b_ffn1  = (const float*)d_in[14];
  const float* w_ffn2  = (const float*)d_in[15];
  const float* b_ffn2  = (const float*)d_in[16];
  const float* ln2_w   = (const float*)d_in[17];
  const float* ln2_b   = (const float*)d_in[18];
  const float* w_cls   = (const float*)d_in[19];
  const float* b_cls   = (const float*)d_in[20];

  float* ws = (float*)d_ws;
  size_t o = 0;
  float* v      = ws + o; o += (size_t)BS_ * C_;    // 160 MB projected values
  float* offr   = ws + o; o += (size_t)BQ_ * 256;   // sampling offsets (raw)
  float* logit  = ws + o; o += (size_t)BQ_ * 128;   // attn logits
  float* attnw  = ws + o; o += (size_t)BQ_ * 128;   // attn weights
  float* feat   = ws + o; o += (size_t)BQ_ * C_;    // sampled per-head features
  float* featp  = ws + o; o += (size_t)BQ_ * C_;    // after out-proj
  float* x1     = ws + o; o += (size_t)BQ_ * C_;    // after LN1
  float* hh     = ws + o; o += (size_t)BQ_ * FFN_;  // FFN hidden
  float* ho     = ws + o; o += (size_t)BQ_ * C_;    // FFN out
  float* x2     = ws + o; o += (size_t)BQ_ * C_;    // after LN2

  o = (o + 7) & ~(size_t)7;                         // 32B-align bf16 arena
  __bf16* bfw = (__bf16*)(ws + o);
  size_t bo = 0;
  __bf16* wv_p    = bfw + bo; bo += (size_t)C_ * C_;
  __bf16* woff_p  = bfw + bo; bo += (size_t)C_ * 256;
  __bf16* wattn_p = bfw + bo; bo += (size_t)C_ * 128;
  __bf16* wout_p  = bfw + bo; bo += (size_t)C_ * C_;
  __bf16* wffn1_p = bfw + bo; bo += (size_t)C_ * FFN_;
  __bf16* wffn2_p = bfw + bo; bo += (size_t)FFN_ * C_;

  const dim3 blk(256);

  // 0. pre-pack all GEMM weights into fragment-native bf16 layout (one-time ~1.4MB)
  pack_w_bf16<<<dim3((C_ * C_ + 255) / 256),   blk, 0, stream>>>(w_value, wv_p,    C_,  C_);
  pack_w_bf16<<<dim3((C_ * 256 + 255) / 256),  blk, 0, stream>>>(w_off,   woff_p,  C_,  256);
  pack_w_bf16<<<dim3((C_ * 128 + 255) / 256),  blk, 0, stream>>>(w_attn,  wattn_p, C_,  128);
  pack_w_bf16<<<dim3((C_ * C_ + 255) / 256),   blk, 0, stream>>>(w_out,   wout_p,  C_,  C_);
  pack_w_bf16<<<dim3((C_ * FFN_ + 255) / 256), blk, 0, stream>>>(w_ffn1,  wffn1_p, C_,  FFN_);
  pack_w_bf16<<<dim3((FFN_ * C_ + 255) / 256), blk, 0, stream>>>(w_ffn2,  wffn2_p, FFN_, C_);

  // 1. value projection: [BS,256] @ [256,256]   (dominant, HBM-streaming)
  gemm_bf16_wmma<<<dim3(BS_ / 32, 1), blk, 0, stream>>>(
      value, wv_p, b_value, v, BS_, C_, C_, 0, /*cg=*/4);

  // 2. sampling offsets: [BQ,256] @ [256,256]
  gemm_bf16_wmma<<<dim3(BQ_ / 32, 1), blk, 0, stream>>>(
      query, woff_p, b_off, offr, BQ_, C_, 256, 0, 4);

  // 3. attention logits: [BQ,256] @ [256,128]
  gemm_bf16_wmma<<<dim3((BQ_ + 63) / 64, 1), blk, 0, stream>>>(
      query, wattn_p, b_attn, logit, BQ_, C_, 128, 0, 2);

  // 4. softmax over L*P per head
  softmax_lp<<<dim3(BQ_), dim3(128), 0, stream>>>(logit, attnw);

  // 5. deformable bilinear sampling (one wave per (b,q,m), lane = channel)
  deform_sample<<<dim3(BQ_), blk, 0, stream>>>(v, refp, offr, attnw, feat);

  // 6. output projection
  gemm_bf16_wmma<<<dim3(BQ_ / 32, 1), blk, 0, stream>>>(
      feat, wout_p, b_out, featp, BQ_, C_, C_, 0, 4);

  // 7. x1 = LN(query + featp)
  add_layernorm<<<dim3(BQ_ / 8), blk, 0, stream>>>(query, featp, ln1_w, ln1_b, x1, BQ_);

  // 8. FFN1 + ReLU: [BQ,256] @ [256,1024]
  gemm_bf16_wmma<<<dim3(BQ_ / 16, 2), blk, 0, stream>>>(
      x1, wffn1_p, b_ffn1, hh, BQ_, C_, FFN_, 1, /*cg=*/8);

  // 9. FFN2: [BQ,1024] @ [1024,256]
  gemm_bf16_wmma<<<dim3(BQ_ / 32, 1), blk, 0, stream>>>(
      hh, wffn2_p, b_ffn2, ho, BQ_, FFN_, C_, 0, 4);

  // 10. x2 = LN(x1 + ho)
  add_layernorm<<<dim3(BQ_ / 8), blk, 0, stream>>>(x1, ho, ln2_w, ln2_b, x2, BQ_);

  // 11. classification head -> d_out [BQ,10]
  cls_head<<<dim3((BQ_ * NC_ + 255) / 256), blk, 0, stream>>>(
      x2, w_cls, b_cls, (float*)d_out);
}
